// BPCheckNode_86260123173236
// MI455X (gfx1250) — matrix operand order repaired and checked by reading, not compiled, not run on hardware
//
#include <hip/hip_runtime.h>
#include <hip/hip_bf16.h>
#include <math.h>

// Problem constants (from reference): B, NE, ND, E, H
#define B_   4
#define NE_  50000
#define ND_  25000
#define E_   200000
#define H_   128

#define LLR_CLAMP  15.0f
#define TANH_CLAMP 0.9999f

typedef __attribute__((ext_vector_type(16))) _Float16 v16h;
typedef __attribute__((ext_vector_type(8)))  float    v8f;
typedef __attribute__((ext_vector_type(2)))  float    v2f;

#if __has_builtin(__builtin_amdgcn_wmma_f32_16x16x4_f32)
#define HAVE_WMMA_F32X4 1
#endif

// ---------------------------------------------------------------------------
// Workspace layout (floats):
//   neg_count  : B*ND
//   logabs_sum : B*ND
//   sign_pe    : B*E
//   logabs_pe  : B*E
//   bp_agg     : B*NE
//   count_e    : NE
// total ~2.05M floats (~8.2 MB)
// ---------------------------------------------------------------------------

__global__ void zero_k(float* __restrict__ p, int n) {
    int i = blockIdx.x * blockDim.x + threadIdx.x;
    if (i < n) p[i] = 0.0f;
}

__global__ void count_k(const int* __restrict__ e_dst, float* __restrict__ count_e) {
    int e = blockIdx.x * blockDim.x + threadIdx.x;
    if (e < E_) atomicAdd(&count_e[e_dst[e]], 1.0f);
}

// ---------------------------------------------------------------------------
// Pass 1: per (b, 16-edge group) wave.
//   llr = hE[b, e_src[e], :] . w_llr  via 4 chained v_wmma_f32_16x16x32_f16
//   then tanh/sign/log and scatter-add into per-(b,d) accumulators.
// A layout (16-bit 16x32): lanes 0-15 row M=lane, K = {0..7, 16..23};
//                          lanes 16-31 row M=lane-16, K = {8..15, 24..31}.
// B holds w_llr replicated across all 16 columns -> every D column = dot.
// ---------------------------------------------------------------------------
__global__ void __launch_bounds__(128)
pass1_k(const float* __restrict__ hE,
        const float* __restrict__ w_llr,
        const float* __restrict__ b_llr,
        const int*   __restrict__ e_src,
        const int*   __restrict__ d_dst,
        float* __restrict__ neg_count,
        float* __restrict__ logabs_sum,
        float* __restrict__ sign_pe,
        float* __restrict__ logabs_pe) {
    __shared__ float smem[4][16];

    const int tid    = blockIdx.x * blockDim.x + threadIdx.x;
    const int wave   = tid >> 5;
    const int lane   = threadIdx.x & 31;
    const int wv     = threadIdx.x >> 5;
    const int groups = E_ / 16;                 // 12500
    const int b      = wave / groups;
    const int g      = wave % groups;
    const int e0     = g * 16;
    const int m16    = lane & 15;
    const int hi     = lane >> 4;

    const int row_idx = e_src[e0 + m16];
    const float* __restrict__ row = hE + ((size_t)b * NE_ + row_idx) * H_;

    v8f c = {0.f, 0.f, 0.f, 0.f, 0.f, 0.f, 0.f, 0.f};

    const int ko = hi * 8;       // A: K sub-offset for this half-wave
    const int kb = hi * 16;      // unused: B indexes with ko, same k order as A

    #pragma unroll
    for (int ch = 0; ch < 4; ++ch) {
        const int cb = ch * 32;
        v16h a, bm;
        #pragma unroll
        for (int i = 0; i < 8; ++i) {
            a[i]     = (_Float16)row[cb + ko + i];        // K = cb+ko..+7
            a[8 + i] = (_Float16)row[cb + 16 + ko + i];   // K = cb+16+ko..+7
        }
        #pragma unroll
        for (int h = 0; h < 16; ++h) {
            bm[h] = (_Float16)w_llr[cb + ko + (h & 7) + (h >> 3) * 16]; // B[k][n] = w[k], all n
        }
        c = __builtin_amdgcn_wmma_f32_16x16x32_f16(
                false, a, false, bm, (short)0, c, false, false);
    }

    // Extract column 0 of D: lane 0 holds rows 0-7, lane 16 holds rows 8-15.
    if (lane == 0) {
        #pragma unroll
        for (int r = 0; r < 8; ++r) smem[wv][r] = c[r];
    }
    if (lane == 16) {
        #pragma unroll
        for (int r = 0; r < 8; ++r) smem[wv][8 + r] = c[r];
    }
    __syncthreads();

    if (lane < 16) {
        const int e  = e0 + lane;
        float llr = smem[wv][lane] + b_llr[0];
        llr = fminf(fmaxf(llr, -LLR_CLAMP), LLR_CLAMP);
        float t = tanhf(0.5f * llr);
        t = fminf(fmaxf(t, -TANH_CLAMP), TANH_CLAMP);
        float s  = (t > 0.f) ? 1.f : ((t < 0.f) ? -1.f : 0.f);
        float la = logf(fmaxf(fabsf(t), 1e-30f));

        sign_pe[(size_t)b * E_ + e]   = s;
        logabs_pe[(size_t)b * E_ + e] = la;

        const int d = d_dst[e];
        if (s < 0.f) atomicAdd(&neg_count[b * ND_ + d], 1.0f);
        atomicAdd(&logabs_sum[b * ND_ + d], la);
    }
}

// ---------------------------------------------------------------------------
// Pass 2: per (b, e) thread — extrinsic message + scatter to variable nodes.
// ---------------------------------------------------------------------------
__global__ void pass2_k(const float* __restrict__ syndrome,
                        const int*   __restrict__ d_dst,
                        const int*   __restrict__ e_dst,
                        const float* __restrict__ neg_count,
                        const float* __restrict__ logabs_sum,
                        const float* __restrict__ sign_pe,
                        const float* __restrict__ logabs_pe,
                        float* __restrict__ bp_agg) {
    int tid = blockIdx.x * blockDim.x + threadIdx.x;
    if (tid >= B_ * E_) return;
    int b = tid / E_;
    int e = tid - b * E_;

    int d = d_dst[e];
    float s   = sign_pe[tid];
    float la  = logabs_pe[tid];
    float nc  = neg_count[b * ND_ + d];
    float sp  = 1.f - 2.f * fmodf(nc, 2.f);
    float ls  = logabs_sum[b * ND_ + d];
    float ss  = 1.f - 2.f * syndrome[b * ND_ + d];

    float se  = sp * s * ss;
    float et  = se * expf(ls - la);
    et = fminf(fmaxf(et, -TANH_CLAMP), TANH_CLAMP);
    float msg = logf((1.f + et) / (1.f - et));   // 2*atanh(et)
    msg = fminf(fmaxf(msg, -LLR_CLAMP), LLR_CLAMP);

    atomicAdd(&bp_agg[b * NE_ + e_dst[e]], msg);
}

// ---------------------------------------------------------------------------
// Projection: out[b,e,h] = (bp_agg[b,e]/max(count,1)) * w_proj[h] + b_proj[h]
// Rank-1 outer product per 16x16 tile via v_wmma_f32_16x16x4_f32 (K=0 only).
// D layout: VGPR r -> row r (lanes 0-15) / row r+8 (lanes 16-31), N = lane&15.
// ---------------------------------------------------------------------------
__global__ void __launch_bounds__(128)
proj_k(const float* __restrict__ bp_agg,
       const float* __restrict__ count_e,
       const float* __restrict__ w_proj,
       const float* __restrict__ b_proj,
       float* __restrict__ out) {
    const int tid    = blockIdx.x * blockDim.x + threadIdx.x;
    const int wave   = tid >> 5;
    const int lane   = threadIdx.x & 31;
    const int groups = NE_ / 16;                // 3125
    const int b      = wave / groups;
    const int g      = wave % groups;
    const int e0     = g * 16;
    const int m16    = lane & 15;
    const int hi     = lane >> 4;

    float bp  = bp_agg[(size_t)b * NE_ + e0 + m16];
    float cnt = fmaxf(count_e[e0 + m16], 1.0f);
    float val = bp / cnt;

#ifdef HAVE_WMMA_F32X4
    // A (16x4): only K=0 column nonzero -> lanes 0-15, VGPR0.
    v2f a;
    a.x = hi ? 0.f : val;
    a.y = 0.f;

    #pragma unroll
    for (int t = 0; t < 8; ++t) {
        // B (4x16): only K=0 row nonzero -> lanes 0-15, VGPR0.
        v2f bb;
        bb.x = hi ? 0.f : w_proj[t * 16 + m16];
        bb.y = 0.f;

        float cj = b_proj[t * 16 + m16];        // broadcast down all 16 rows
        v8f cc = {cj, cj, cj, cj, cj, cj, cj, cj};

        v8f dmat = __builtin_amdgcn_wmma_f32_16x16x4_f32(
                       false, a, false, bb, (short)0, cc, false, false);

        #pragma unroll
        for (int r = 0; r < 8; ++r) {
            const int rowm = r + 8 * hi;
            out[((size_t)b * NE_ + e0 + rowm) * H_ + t * 16 + m16] = dmat[r];
        }
    }
#else
    // VALU fallback: broadcast each row's value via shuffle, fma per column.
    #pragma unroll 1
    for (int r = 0; r < 16; ++r) {
        float vr = __shfl(val, r, 32);
        const size_t base = ((size_t)b * NE_ + e0 + r) * H_;
        #pragma unroll
        for (int j = 0; j < 4; ++j) {
            int col = lane + 32 * j;
            out[base + col] = fmaf(vr, w_proj[col], b_proj[col]);
        }
    }
#endif
}

// ---------------------------------------------------------------------------
extern "C" void kernel_launch(void* const* d_in, const int* in_sizes, int n_in,
                              void* d_out, int out_size, void* d_ws, size_t ws_size,
                              hipStream_t stream) {
    (void)in_sizes; (void)n_in; (void)out_size; (void)ws_size;

    const float* hE       = (const float*)d_in[0];
    /* d_in[1] = hD : unused by the reference computation */
    const float* syndrome = (const float*)d_in[2];
    const int*   edge_e2d = (const int*)d_in[3];
    const int*   edge_d2e = (const int*)d_in[4];
    const float* w_llr    = (const float*)d_in[5];
    const float* b_llr    = (const float*)d_in[6];
    const float* w_proj   = (const float*)d_in[7];
    const float* b_proj   = (const float*)d_in[8];
    float*       out      = (float*)d_out;

    float* ws         = (float*)d_ws;
    float* neg_count  = ws;                          // B*ND
    float* logabs_sum = ws + (size_t)B_ * ND_;       // B*ND
    float* sign_pe    = ws + (size_t)2 * B_ * ND_;   // B*E
    float* logabs_pe  = sign_pe + (size_t)B_ * E_;   // B*E
    float* bp_agg     = logabs_pe + (size_t)B_ * E_; // B*NE
    float* count_e    = bp_agg + (size_t)B_ * NE_;   // NE

    const int* e_src = edge_e2d;        // edge_e2d[0]
    const int* d_dst = edge_e2d + E_;   // edge_e2d[1]
    const int* e_dst = edge_d2e + E_;   // edge_d2e[1]

    // Zero accumulators (two contiguous ranges).
    {
        int n0 = 2 * B_ * ND_;                  // neg_count + logabs_sum
        zero_k<<<(n0 + 255) / 256, 256, 0, stream>>>(neg_count, n0);
        int n1 = B_ * NE_ + NE_;                // bp_agg + count_e
        zero_k<<<(n1 + 255) / 256, 256, 0, stream>>>(bp_agg, n1);
    }

    // Edge-degree counts for variable nodes.
    count_k<<<(E_ + 255) / 256, 256, 0, stream>>>(e_dst, count_e);

    // Pass 1: WMMA dot + check-node scatter.  waves = B*(E/16) = 50000.
    pass1_k<<<(B_ * (E_ / 16)) / 4, 128, 0, stream>>>(
        hE, w_llr, b_llr, e_src, d_dst,
        neg_count, logabs_sum, sign_pe, logabs_pe);

    // Pass 2: extrinsic messages + variable-node scatter.
    pass2_k<<<(B_ * E_ + 255) / 256, 256, 0, stream>>>(
        syndrome, d_dst, e_dst, neg_count, logabs_sum,
        sign_pe, logabs_pe, bp_agg);

    // Projection: WMMA rank-1 outer product, writes the 102 MB output.
    proj_k<<<(B_ * (NE_ / 16)) / 4, 128, 0, stream>>>(
        bp_agg, count_e, w_proj, b_proj, out);
}
